// Qlora_4_bit_52853867544998
// MI455X (gfx1250) — matrix operand order repaired and checked by reading, compiled, not run on hardware
//
#include <hip/hip_runtime.h>
#include <hip/hip_bf16.h>

// ---------------------------------------------------------------------------
// QLoRA 4-bit forward for MI455X (gfx1250, wave32, WMMA)
//   y = x @ dequant(W)^T + 2.0 * (x @ W_a @ W_b)
// K1: fold NF4 dequant + rank-16 LoRA into W_eff (f16) in d_ws
// K2: 128x128 block / 64 K-step double-buffered WMMA GEMM
//     - B tile staged via GLOBAL_LOAD_ASYNC_TO_LDS_B128 (ASYNCcnt path)
//     - A tile prefetched to regs, cvt fp32->f16, ds_store
//     - v_wmma_f32_16x16x32_f16, 16 WMMA per wave per LDS tile
// ---------------------------------------------------------------------------

typedef __attribute__((ext_vector_type(16))) _Float16 v16h;
typedef __attribute__((ext_vector_type(8)))  _Float16 v8h;
typedef __attribute__((ext_vector_type(4)))  _Float16 v4h;
typedef __attribute__((ext_vector_type(8)))  float    v8f;

#define M_DIM 8192   // B*S = 4*2048
#define K_DIM 4096   // IN
#define N_DIM 4096   // OUT
#define QBLK  64     // NF4 blocksize
#define LORA_R 16
#define LORA_SCALE 2.0f  // alpha/r = 32/16

__constant__ float NF4_CODE[16] = {
    -1.0f, -0.6961928009986877f, -0.5250730514526367f, -0.39491748809814453f,
    -0.28444138169288635f, -0.18477343022823334f, -0.09105003625154495f, 0.0f,
    0.07958029955625534f, 0.16093020141124725f, 0.24611230194568634f,
    0.33791524171829224f, 0.44070982933044434f, 0.5626170039176941f,
    0.7229568362236023f, 1.0f};

// --- CDNA5 async global->LDS (ISA 15.18.3 op 98) + ASYNCcnt wait -----------
__device__ __forceinline__ void async_b128_to_lds(unsigned lds_byte_addr,
                                                  const void* gaddr) {
    asm volatile("global_load_async_to_lds_b128 %0, %1, off"
                 :: "v"(lds_byte_addr), "v"(gaddr) : "memory");
}
__device__ __forceinline__ void wait_asynccnt0() {
    asm volatile("s_wait_asynccnt 0" ::: "memory");
}
__device__ __forceinline__ unsigned lds_addr_of(const void* p) {
    // ISA 10.2: flat address of LDS maps to LDS via low 32 bits
    return (unsigned)(unsigned long long)p;
}

// ---------------------------------------------------------------------------
// Kernel 1: dequantize NF4 + fold rank-16 LoRA into W_eff (f16, [N_DIM][K_DIM])
// ---------------------------------------------------------------------------
__global__ __launch_bounds__(256) void dequant_fold_kernel(
    const int*   __restrict__ w_codes,
    const float* __restrict__ absmax,
    const float* __restrict__ Wa,        // [K_DIM][LORA_R]
    const float* __restrict__ Wb,        // [LORA_R][N_DIM]
    _Float16*    __restrict__ Weff)      // [N_DIM][K_DIM]
{
    const int o  = blockIdx.y;
    const int i0 = (blockIdx.x * 256 + threadIdx.x) * 4;

    __shared__ __align__(16) float wb_sm[LORA_R];
    if (threadIdx.x < LORA_R) wb_sm[threadIdx.x] = Wb[threadIdx.x * N_DIM + o];
    __syncthreads();

    float wbr[LORA_R];
#pragma unroll
    for (int r = 0; r < LORA_R; ++r) wbr[r] = wb_sm[r];

    const int4  c  = *(const int4*)(w_codes + (size_t)o * K_DIM + i0);
    const float am = absmax[o * (K_DIM / QBLK) + (i0 >> 6)];

    float wq[4];
    wq[0] = NF4_CODE[c.x & 15] * am;
    wq[1] = NF4_CODE[c.y & 15] * am;
    wq[2] = NF4_CODE[c.z & 15] * am;
    wq[3] = NF4_CODE[c.w & 15] * am;

    v4h outv;
#pragma unroll
    for (int j = 0; j < 4; ++j) {
        const float4* wa = (const float4*)(Wa + (size_t)(i0 + j) * LORA_R);
        float acc = 0.f;
#pragma unroll
        for (int q = 0; q < 4; ++q) {
            const float4 v = wa[q];
            acc += v.x * wbr[4 * q + 0] + v.y * wbr[4 * q + 1]
                 + v.z * wbr[4 * q + 2] + v.w * wbr[4 * q + 3];
        }
        outv[j] = (_Float16)(wq[j] + LORA_SCALE * acc);
    }
    *(v4h*)(Weff + (size_t)o * K_DIM + i0) = outv;
}

// ---------------------------------------------------------------------------
// Kernel 2: double-buffered WMMA GEMM
// ---------------------------------------------------------------------------
#define TM 128
#define TN 128
#define TK 64
#define LDA 72   // padded row (halfs): 144B stride -> conflict-free b128

__global__ __launch_bounds__(256) void qlora_wmma_gemm(
    const float*    __restrict__ X,      // [M_DIM][K_DIM] fp32
    const _Float16* __restrict__ Weff,   // [N_DIM][K_DIM] f16
    float*          __restrict__ Y)      // [M_DIM][N_DIM] fp32
{
    __shared__ _Float16 As[2][TM * LDA];   // 2 x 18 KB
    __shared__ _Float16 Bs[2][TN * LDA];   // 2 x 18 KB

    const int tid  = threadIdx.x;
    const int wave = tid >> 5;
    const int lane = tid & 31;
    const int wm   = wave & 1;    // 2 wave-rows -> 64 M each
    const int wn   = wave >> 1;   // 4 wave-cols -> 32 N each
    const int m0   = blockIdx.y * TM;
    const int n0   = blockIdx.x * TN;
    const int lh   = lane >> 4;
    const int l15  = lane & 15;

    // staging decomposition
    const int arow = tid >> 4;          // A: 16 rows/pass, 16 thr/row, 4 f32 ea
    const int acol = (tid & 15) * 4;
    const int brow = tid >> 3;          // B: 32 rows/pass, 8 thr/row, 8 f16 ea
    const int bcol = (tid & 7) * 8;

    v8f acc[4][2];
    const v8f vzero = {0.f, 0.f, 0.f, 0.f, 0.f, 0.f, 0.f, 0.f};
#pragma unroll
    for (int i = 0; i < 4; ++i)
#pragma unroll
        for (int j = 0; j < 2; ++j) acc[i][j] = vzero;

    float4 apf[8];

    // ---- prologue: stage tile 0 into buffer 0 ----
#pragma unroll
    for (int p = 0; p < 4; ++p) {
        const int row = brow + p * 32;
        async_b128_to_lds(lds_addr_of(&Bs[0][row * LDA + bcol]),
                          Weff + (size_t)(n0 + row) * K_DIM + bcol);
    }
#pragma unroll
    for (int p = 0; p < 8; ++p)
        apf[p] = *(const float4*)(X + (size_t)(m0 + arow + p * 16) * K_DIM + acol);
#pragma unroll
    for (int p = 0; p < 8; ++p) {
        v4h h;
        h[0] = (_Float16)apf[p].x; h[1] = (_Float16)apf[p].y;
        h[2] = (_Float16)apf[p].z; h[3] = (_Float16)apf[p].w;
        *(v4h*)&As[0][(arow + p * 16) * LDA + acol] = h;
    }
    wait_asynccnt0();
    __syncthreads();

    int buf = 0;
    for (int k0 = 0; k0 < K_DIM; k0 += TK) {
        const int  nxt = buf ^ 1;
        const bool pf  = (k0 + TK) < K_DIM;

        // ---- prefetch tile k0+TK while computing on `buf` ----
        if (pf) {
            const int kn = k0 + TK;
#pragma unroll
            for (int p = 0; p < 4; ++p) {
                const int row = brow + p * 32;
                async_b128_to_lds(lds_addr_of(&Bs[nxt][row * LDA + bcol]),
                                  Weff + (size_t)(n0 + row) * K_DIM + kn + bcol);
            }
#pragma unroll
            for (int p = 0; p < 8; ++p)
                apf[p] = *(const float4*)(X + (size_t)(m0 + arow + p * 16) * K_DIM + kn + acol);
        }

        // ---- compute: 2 sub-steps of K=32, 8 WMMA each ----
#pragma unroll
        for (int kk = 0; kk < TK; kk += 32) {
            v16h afrag[4];
#pragma unroll
            for (int mt = 0; mt < 4; ++mt) {
                const int row  = wm * 64 + mt * 16 + l15;
                const int kofs = kk + lh * 8;
                const v8h lo = *(const v8h*)&As[buf][row * LDA + kofs];
                const v8h hi = *(const v8h*)&As[buf][row * LDA + kofs + 16];
#pragma unroll
                for (int e = 0; e < 8; ++e) { afrag[mt][e] = lo[e]; afrag[mt][e + 8] = hi[e]; }
            }
            v16h bfrag[2];
#pragma unroll
            for (int nt = 0; nt < 2; ++nt) {
                const int row  = wn * 32 + nt * 16 + l15;
                const int kofs = kk + lh * 8;
                const v8h lo = *(const v8h*)&Bs[buf][row * LDA + kofs];
                const v8h hi = *(const v8h*)&Bs[buf][row * LDA + kofs + 16];
#pragma unroll
                for (int e = 0; e < 8; ++e) { bfrag[nt][e] = lo[e]; bfrag[nt][e + 8] = hi[e]; }
            }
#pragma unroll
            for (int mt = 0; mt < 4; ++mt)
#pragma unroll
                for (int nt = 0; nt < 2; ++nt)
                    acc[mt][nt] = __builtin_amdgcn_wmma_f32_16x16x32_f16(
                        false, afrag[mt], false, bfrag[nt],
                        (short)0, acc[mt][nt], false, false);
        }

        // ---- drain prefetch into `nxt` buffer ----
        if (pf) {
#pragma unroll
            for (int p = 0; p < 8; ++p) {
                v4h h;
                h[0] = (_Float16)apf[p].x; h[1] = (_Float16)apf[p].y;
                h[2] = (_Float16)apf[p].z; h[3] = (_Float16)apf[p].w;
                *(v4h*)&As[nxt][(arow + p * 16) * LDA + acol] = h;
            }
        }
        wait_asynccnt0();
        __syncthreads();
        buf = nxt;
    }

    // ---- epilogue: VGPR j -> M = (lane<16 ? j : 8+j), N = lane&15 ----
#pragma unroll
    for (int mt = 0; mt < 4; ++mt) {
#pragma unroll
        for (int nt = 0; nt < 2; ++nt) {
            const int col   = n0 + wn * 32 + nt * 16 + l15;
            const int rbase = m0 + wm * 64 + mt * 16 + lh * 8;
#pragma unroll
            for (int j = 0; j < 8; ++j)
                Y[(size_t)(rbase + j) * N_DIM + col] = acc[mt][nt][j];
        }
    }
}

// ---------------------------------------------------------------------------
extern "C" void kernel_launch(void* const* d_in, const int* in_sizes, int n_in,
                              void* d_out, int out_size, void* d_ws, size_t ws_size,
                              hipStream_t stream) {
    (void)in_sizes; (void)n_in; (void)out_size; (void)ws_size;
    const float* x       = (const float*)d_in[0];
    const int*   w_codes = (const int*)  d_in[1];
    const float* absmax  = (const float*)d_in[2];
    const float* Wa      = (const float*)d_in[3];
    const float* Wb      = (const float*)d_in[4];
    float*       y       = (float*)d_out;
    _Float16*    Weff    = (_Float16*)d_ws;   // 32 MB scratch

    dim3 g1(K_DIM / 1024, N_DIM);
    dequant_fold_kernel<<<g1, 256, 0, stream>>>(w_codes, absmax, Wa, Wb, Weff);

    dim3 g2(N_DIM / TN, M_DIM / TM);
    qlora_wmma_gemm<<<g2, 256, 0, stream>>>(x, Weff, y);
}
